// MultiHeadAttention_76725295776293
// MI455X (gfx1250) — matrix operand order repaired and checked by reading, not compiled, not run on hardware
//
#include <hip/hip_runtime.h>

// Problem constants
#define Bz  4
#define Tz  2048
#define Dz  1024
#define Hz  16
#define HDz 64

typedef __bf16 v16bf __attribute__((ext_vector_type(16)));
typedef float  v8f   __attribute__((ext_vector_type(8)));
typedef unsigned int u32x4 __attribute__((ext_vector_type(4)));
typedef int  i32x4 __attribute__((ext_vector_type(4)));
typedef int  i32x8 __attribute__((ext_vector_type(8)));

static __device__ __forceinline__ v8f wmma_bf16(v16bf a, v16bf b, v8f c) {
  // D = A(16x32 bf16) * B(32x16 bf16) + C(16x16 f32)
  return __builtin_amdgcn_wmma_f32_16x16x32_bf16(false, a, false, b, (short)0, c,
                                                 false, false);
}

// A-fragment (16x32 bf16) gather from a row-major [16 x >=32] buffer, stride ld.
// Element e of v16bf -> K = 8*(lane>>4) + (e&7) + 16*(e>>3): per-lane this is two
// contiguous 16B runs, so it lowers to paired ds_load_b128 / global_load_b128.
static __device__ __forceinline__ v16bf load_a_frag(const __bf16* __restrict__ p, int ld) {
  const int l  = (int)(threadIdx.x & 31u);
  const int m  = l & 15;
  const int kb = (l >> 4) << 3;
  v16bf a;
#pragma unroll
  for (int e = 0; e < 16; ++e) {
    const int k = kb + (e & 7) + ((e >> 3) << 4);
    a[e] = p[m * ld + k];
  }
  return a;
}

// CDNA5 async copy: 16 bytes per lane, global -> LDS, tracked by ASYNCcnt.
static __device__ __forceinline__ void async_ld16(void* lds, const void* g) {
  const unsigned lo = (unsigned)(size_t)lds;
  const unsigned long long ga = (unsigned long long)(size_t)g;
  asm volatile("global_load_async_to_lds_b128 %0, %1, off"
               :: "v"(lo), "v"(ga)
               : "memory");
}
static __device__ __forceinline__ void wait_async0() {
  asm volatile("s_wait_asynccnt 0x0" ::: "memory");
}

// CDNA5 Tensor Data Mover: DMA a 2D bf16 tile (tile_w x tile_h elems) from a
// row-major tensor (row pitch row_stride elems) into LDS, with hardware row
// padding. Descriptor layout per CDNA5 ISA.
static __device__ __forceinline__ void tdm_load_2d(void* lds, const void* gsrc,
                                                   unsigned tile_w, unsigned tile_h,
                                                   unsigned tensor_w, unsigned tensor_h,
                                                   unsigned row_stride,
                                                   unsigned pad_interval_code,
                                                   unsigned pad_amount_code) {
  const unsigned long long ga = (unsigned long long)(size_t)gsrc;
  const unsigned lo = (unsigned)(size_t)lds;
  u32x4 g0;
  g0[0] = 1u;                                               // count=1 (valid user D#)
  g0[1] = lo;                                               // lds_addr (bytes)
  g0[2] = (unsigned)(ga & 0xffffffffu);                     // global_addr[31:0]
  g0[3] = (unsigned)((ga >> 32) & 0x01ffffffu) | (2u << 30);// global_addr[56:32]|type=2
  i32x8 g1;
  g1[0] = (int)((1u << 16)                                  // data_size = 2 bytes
              | (1u << 20)                                  // pad_enable
              | (pad_interval_code << 22)
              | (pad_amount_code << 25));
  g1[1] = (int)((tensor_w & 0xffffu) << 16);                // tensor_dim0[15:0]
  g1[2] = (int)(((tensor_w >> 16) & 0xffffu)
              | ((tensor_h & 0xffffu) << 16));              // dim0[31:16] | dim1[15:0]
  g1[3] = (int)(((tensor_h >> 16) & 0xffffu)
              | ((tile_w & 0xffffu) << 16));                // dim1[31:16] | tile_dim0
  g1[4] = (int)(tile_h & 0xffffu);                          // tile_dim1 (tile_dim2=0)
  g1[5] = (int)row_stride;                                  // tensor_dim0_stride[31:0]
  g1[6] = 0;
  g1[7] = 0;
  i32x4 z4 = {0, 0, 0, 0};
#if defined(__clang_major__) && (__clang_major__ >= 23)
  i32x8 z8 = {0, 0, 0, 0, 0, 0, 0, 0};
  __builtin_amdgcn_tensor_load_to_lds(g0, g1, z4, z4, z8, 0);
#else
  __builtin_amdgcn_tensor_load_to_lds(g0, g1, z4, z4, 0);
#endif
}

// ---------------------------------------------------------------------------
// Kernel 1: qkv = x @ W_qkv + b_qkv  (fp32 in, bf16 out, laid out [b][3][h][t][64])
// 256 threads = 8 waves (2x4); wave tile 32x32 (2x2 WMMA); block tile 64x128;
// K chunk 64 -> 8 WMMAs per stage. Software-pipelined: global loads for chunk
// i+1 issue before computing chunk i; cvt+LDS store after; one barrier/iter.
// ---------------------------------------------------------------------------
__global__ __launch_bounds__(256) void qkv_gemm(const float* __restrict__ x,
                                                const float* __restrict__ Wq,
                                                const float* __restrict__ bq,
                                                __bf16* __restrict__ qkvws) {
  __shared__ __align__(16) __bf16 As[2][64 * 72];    // 64 rows x 64 K (pad 72)
  __shared__ __align__(16) __bf16 Bs[2][64 * 144];   // 64 K x 128 N (pad 144)
  const int tid = threadIdx.x;
  const int l = tid & 31, w = tid >> 5;
  const int wr = w >> 2, wc = w & 3;
  const int m0 = wr * 32, n0 = wc * 32;
  const int row0 = blockIdx.y * 64;
  const int col0 = blockIdx.x * 128;
  v8f acc[2][2];
#pragma unroll
  for (int i = 0; i < 2; ++i)
#pragma unroll
    for (int j = 0; j < 2; ++j) { v8f z = {}; acc[i][j] = z; }

  const int ar = tid >> 2, ac0 = (tid & 3) << 4;   // A stage: 16 fp32/thread
  const int br = tid >> 2, bc0 = (tid & 3) << 5;   // B stage: 32 fp32/thread
  const float* aptr = x + (size_t)(row0 + ar) * Dz + ac0;
  const float* bptr = Wq + (size_t)br * (3 * Dz) + col0 + bc0;

  float4 areg[4], breg[8];
  auto store_stage = [&](int buf) {
    __bf16* da = &As[buf][ar * 72 + ac0];
#pragma unroll
    for (int i = 0; i < 4; ++i) {
      da[i * 4 + 0] = (__bf16)areg[i].x; da[i * 4 + 1] = (__bf16)areg[i].y;
      da[i * 4 + 2] = (__bf16)areg[i].z; da[i * 4 + 3] = (__bf16)areg[i].w;
    }
    __bf16* db = &Bs[buf][br * 144 + bc0];
#pragma unroll
    for (int i = 0; i < 8; ++i) {
      db[i * 4 + 0] = (__bf16)breg[i].x; db[i * 4 + 1] = (__bf16)breg[i].y;
      db[i * 4 + 2] = (__bf16)breg[i].z; db[i * 4 + 3] = (__bf16)breg[i].w;
    }
  };

  // Prologue: stage chunk 0 into buffer 0
#pragma unroll
  for (int i = 0; i < 4; ++i) areg[i] = ((const float4*)aptr)[i];
#pragma unroll
  for (int i = 0; i < 8; ++i) breg[i] = ((const float4*)bptr)[i];
  store_stage(0);
  __syncthreads();

  for (int it = 0; it < Dz / 64; ++it) {
    const int cur = it & 1, nxt = cur ^ 1;
    if (it < Dz / 64 - 1) {  // issue next chunk's global loads before compute
      const float* a2 = aptr + (it + 1) * 64;
      const float* b2 = bptr + (size_t)(it + 1) * 64 * (3 * Dz);
      __builtin_prefetch(a2 + 64, 0, 3);
      __builtin_prefetch(b2 + (size_t)64 * (3 * Dz), 0, 3);
#pragma unroll
      for (int i = 0; i < 4; ++i) areg[i] = ((const float4*)a2)[i];
#pragma unroll
      for (int i = 0; i < 8; ++i) breg[i] = ((const float4*)b2)[i];
    }
#pragma unroll
    for (int ks = 0; ks < 64; ks += 32) {
      v16bf a0 = load_a_frag(&As[cur][(m0 + 0)  * 72 + ks], 72);
      v16bf a1 = load_a_frag(&As[cur][(m0 + 16) * 72 + ks], 72);
      v16bf b0, b1;
#pragma unroll
      for (int e = 0; e < 16; ++e) {
        const int kk = ks + ((l >> 4) << 3) + (e & 7) + ((e >> 3) << 4);
        b0[e] = Bs[cur][kk * 144 + n0 + (l & 15)];
        b1[e] = Bs[cur][kk * 144 + n0 + 16 + (l & 15)];
      }
      acc[0][0] = wmma_bf16(a0, b0, acc[0][0]);
      acc[0][1] = wmma_bf16(a0, b1, acc[0][1]);
      acc[1][0] = wmma_bf16(a1, b0, acc[1][0]);
      acc[1][1] = wmma_bf16(a1, b1, acc[1][1]);
    }
    if (it < Dz / 64 - 1) store_stage(nxt);
    __syncthreads();
  }

  const int hl = l >> 4, ln = l & 15;
#pragma unroll
  for (int nt = 0; nt < 2; ++nt) {
    const int cg = col0 + n0 + nt * 16 + ln;
    const float bias = bq[cg];
    const int which = cg >> 10;
    const int h = (cg & 1023) >> 6;
    const int d = cg & 63;
#pragma unroll
    for (int mt = 0; mt < 2; ++mt) {
#pragma unroll
      for (int r = 0; r < 8; ++r) {
        const int tg = row0 + m0 + mt * 16 + r + (hl << 3);
        const int bb = tg >> 11, t = tg & 2047;
        qkvws[(size_t)(((bb * 3 + which) * Hz + h) * Tz + t) * HDz + d] =
            (__bf16)(acc[mt][nt][r] + bias);
      }
    }
  }
}

// ---------------------------------------------------------------------------
// Kernel 2: causal flash attention, TDM-pipelined. One block = (b,h,64 q-rows);
// 4 waves x 16 q-rows. Ping-pong K/V LDS buffers: wave 0 issues the TDM for
// tile i+1 right after the barrier that publishes tile i, so the Tensor Data
// Mover streams the next tile while all waves run WMMAs + softmax on this one.
// TDM pad codes: interval 4 (every 32 DWORDs = 64 bf16), amount 7 (+8 DWORDs
// = 16 bf16) -> LDS row pitch 80 elems, matching Kl/Vl.
// ---------------------------------------------------------------------------
__global__ __launch_bounds__(128) void flash_attn(const __bf16* __restrict__ qkv,
                                                  __bf16* __restrict__ attnws) {
  __shared__ __align__(16) __bf16 Kl[2][64 * 80];
  __shared__ __align__(16) __bf16 Vl[2][64 * 80];
  __shared__ __bf16 Pbuf[4 * 16 * 72];
  const int tid = threadIdx.x;
  const int l = tid & 31, w = tid >> 5;
  const int hl = l >> 4, ln = l & 15;
  const int qt = blockIdx.x & 31;
  const int bh = blockIdx.x >> 5;
  const int b = bh >> 4, h = bh & 15;
  const __bf16* qp = qkv + (size_t)((b * 3 + 0) * Hz + h) * Tz * HDz;
  const __bf16* kp = qkv + (size_t)((b * 3 + 1) * Hz + h) * Tz * HDz;
  const __bf16* vp = qkv + (size_t)((b * 3 + 2) * Hz + h) * Tz * HDz;
  const int q0 = qt * 64 + w * 16;

  v16bf qa0 = load_a_frag(qp + (size_t)q0 * HDz, HDz);
  v16bf qa1 = load_a_frag(qp + (size_t)q0 * HDz + 32, HDz);

  v8f o[4];
#pragma unroll
  for (int nc = 0; nc < 4; ++nc) { v8f z = {}; o[nc] = z; }
  float mst[8], lst[8];
#pragma unroll
  for (int r = 0; r < 8; ++r) { mst[r] = -__builtin_inff(); lst[r] = 0.f; }

  __bf16* pw = &Pbuf[w * 16 * 72];

  // Prologue: TDM tile 0 into buffer 0
  if (w == 0) {
    tdm_load_2d(&Kl[0][0], kp, 64, 64, 64, Tz, 64, 4, 7);
    tdm_load_2d(&Vl[0][0], vp, 64, 64, 64, Tz, 64, 4, 7);
  }

  for (int it = 0; it <= qt; ++it) {
    const int cur = it & 1, nxt = cur ^ 1;
    const int kv0 = it * 64;
    if (w == 0) __builtin_amdgcn_s_wait_tensorcnt(0);  // tile it landed in LDS
    __syncthreads();                                   // publish; buf nxt now free
    if (w == 0 && it < qt) {                           // stream tile it+1 under compute
      tdm_load_2d(&Kl[nxt][0], kp + (size_t)(kv0 + 64) * HDz, 64, 64, 64, Tz, 64, 4, 7);
      tdm_load_2d(&Vl[nxt][0], vp + (size_t)(kv0 + 64) * HDz, 64, 64, 64, Tz, 64, 4, 7);
    }
    {  // L2 prefetch two tiles ahead
      const int rr = tid >> 1, cc = (tid & 1) << 5;
      __builtin_prefetch(kp + (size_t)(kv0 + 128 + rr) * HDz + cc, 0, 3);
      __builtin_prefetch(vp + (size_t)(kv0 + 128 + rr) * HDz + cc, 0, 3);
    }
    const __bf16* Kc = &Kl[cur][0];
    const __bf16* Vc = &Vl[cur][0];

    // S = Q @ K^T for 4 key sub-tiles of 16
    v8f s[4];
#pragma unroll
    for (int jt = 0; jt < 4; ++jt) {
      // B frag (k=dim, n=key): lane's key row, dims in table order
      v16bf kb0 = load_a_frag(&Kc[(jt * 16) * 80], 80);
      v16bf kb1 = load_a_frag(&Kc[(jt * 16) * 80 + 32], 80);
      v8f sc = {};
      sc = wmma_bf16(qa0, kb0, sc);
      sc = wmma_bf16(qa1, kb1, sc);
      const int kg = kv0 + jt * 16 + ln;
#pragma unroll
      for (int r = 0; r < 8; ++r) {
        const int qg = q0 + r + (hl << 3);
        s[jt][r] = (kg > qg) ? -__builtin_inff() : sc[r] * 0.125f;
      }
    }

    // Online softmax (rows live in 16-lane halves; shfl_xor masks < 16 stay in-half)
#pragma unroll
    for (int r = 0; r < 8; ++r) {
      float m = fmaxf(fmaxf(s[0][r], s[1][r]), fmaxf(s[2][r], s[3][r]));
#pragma unroll
      for (int off = 1; off < 16; off <<= 1) m = fmaxf(m, __shfl_xor(m, off, 32));
      const float nm  = fmaxf(mst[r], m);
      const float nms = (nm == -__builtin_inff()) ? 0.f : nm;
      const float alpha = (mst[r] == -__builtin_inff()) ? 0.f : __expf(mst[r] - nms);
      mst[r] = nm;
      float rsum = 0.f;
#pragma unroll
      for (int jt = 0; jt < 4; ++jt) {
        const float p = __expf(s[jt][r] - nms);
        s[jt][r] = p;
        rsum += p;
      }
#pragma unroll
      for (int off = 1; off < 16; off <<= 1) rsum += __shfl_xor(rsum, off, 32);
      lst[r] = lst[r] * alpha + rsum;
#pragma unroll
      for (int nc = 0; nc < 4; ++nc) o[nc][r] *= alpha;
#pragma unroll
      for (int jt = 0; jt < 4; ++jt)
        pw[(r + (hl << 3)) * 72 + jt * 16 + ln] = (__bf16)s[jt][r];
    }

    // O += P @ V  (same-wave LDS RAW: compiler inserts s_wait_dscnt)
    v16bf pa0 = load_a_frag(pw, 72);
    v16bf pa1 = load_a_frag(pw + 32, 72);
#pragma unroll
    for (int nc = 0; nc < 4; ++nc) {
      v16bf vb0, vb1;
#pragma unroll
      for (int e = 0; e < 16; ++e) {
        const int kk = ((l >> 4) << 3) + (e & 7) + ((e >> 3) << 4);
        vb0[e] = Vc[kk * 80 + nc * 16 + ln];
        vb1[e] = Vc[(32 + kk) * 80 + nc * 16 + ln];
      }
      o[nc] = wmma_bf16(pa0, vb0, o[nc]);
      o[nc] = wmma_bf16(pa1, vb1, o[nc]);
    }
  }

  // Normalize and emit bf16 attention output as [B*T][D] (head-interleaved cols)
#pragma unroll
  for (int r = 0; r < 8; ++r) {
    const float inv = 1.f / lst[r];
    const int qg = q0 + r + (hl << 3);
    __bf16* dst = attnws + (size_t)(b * Tz + qg) * Dz + h * HDz;
#pragma unroll
    for (int nc = 0; nc < 4; ++nc) dst[nc * 16 + ln] = (__bf16)(o[nc][r] * inv);
  }
}

// ---------------------------------------------------------------------------
// Kernel 3: out = attn @ W_fc + b_fc  (bf16 x fp32 -> fp32). Pipelined like
// kernel 1; A tiles (already bf16) go straight into the next LDS buffer via
// async-to-LDS (ASYNCcnt path) issued before the current chunk's WMMAs.
// ---------------------------------------------------------------------------
__global__ __launch_bounds__(256) void fc_gemm(const __bf16* __restrict__ attn,
                                               const float* __restrict__ Wf,
                                               const float* __restrict__ bfc,
                                               float* __restrict__ out) {
  __shared__ __align__(16) __bf16 As[2][64 * 72];
  __shared__ __align__(16) __bf16 Bs[2][64 * 144];
  const int tid = threadIdx.x;
  const int l = tid & 31, w = tid >> 5;
  const int wr = w >> 2, wc = w & 3;
  const int m0 = wr * 32, n0 = wc * 32;
  const int row0 = blockIdx.y * 64;
  const int col0 = blockIdx.x * 128;
  v8f acc[2][2];
#pragma unroll
  for (int i = 0; i < 2; ++i)
#pragma unroll
    for (int j = 0; j < 2; ++j) { v8f z = {}; acc[i][j] = z; }

  const int br = tid >> 2, bc0 = (tid & 3) << 5;
  const float* bptr = Wf + (size_t)br * Dz + col0 + bc0;

  float4 breg[8];
  auto store_b = [&](int buf) {
    __bf16* db = &Bs[buf][br * 144 + bc0];
#pragma unroll
    for (int i = 0; i < 8; ++i) {
      db[i * 4 + 0] = (__bf16)breg[i].x; db[i * 4 + 1] = (__bf16)breg[i].y;
      db[i * 4 + 2] = (__bf16)breg[i].z; db[i * 4 + 3] = (__bf16)breg[i].w;
    }
  };
  auto async_a = [&](int buf, int k0) {
#pragma unroll
    for (int i = 0; i < 2; ++i) {
      const int u = i * 256 + tid;
      const int rr = u >> 3, cc = (u & 7) << 3;
      async_ld16(&As[buf][rr * 72 + cc], attn + (size_t)(row0 + rr) * Dz + k0 + cc);
    }
  };

  // Prologue: stage chunk 0 into buffer 0
  async_a(0, 0);
#pragma unroll
  for (int i = 0; i < 8; ++i) breg[i] = ((const float4*)bptr)[i];
  store_b(0);
  wait_async0();
  __syncthreads();

  for (int it = 0; it < Dz / 64; ++it) {
    const int cur = it & 1, nxt = cur ^ 1;
    if (it < Dz / 64 - 1) {  // stream next chunk under this chunk's WMMAs
      async_a(nxt, (it + 1) * 64);
      const float* b2 = bptr + (size_t)(it + 1) * 64 * Dz;
      __builtin_prefetch(b2 + (size_t)64 * Dz, 0, 3);
#pragma unroll
      for (int i = 0; i < 8; ++i) breg[i] = ((const float4*)b2)[i];
    }
#pragma unroll
    for (int ks = 0; ks < 64; ks += 32) {
      v16bf a0 = load_a_frag(&As[cur][(m0 + 0)  * 72 + ks], 72);
      v16bf a1 = load_a_frag(&As[cur][(m0 + 16) * 72 + ks], 72);
      v16bf b0, b1;
#pragma unroll
      for (int e = 0; e < 16; ++e) {
        const int kk = ks + ((l >> 4) << 3) + (e & 7) + ((e >> 3) << 4);
        b0[e] = Bs[cur][kk * 144 + n0 + (l & 15)];
        b1[e] = Bs[cur][kk * 144 + n0 + 16 + (l & 15)];
      }
      acc[0][0] = wmma_bf16(a0, b0, acc[0][0]);
      acc[0][1] = wmma_bf16(a0, b1, acc[0][1]);
      acc[1][0] = wmma_bf16(a1, b0, acc[1][0]);
      acc[1][1] = wmma_bf16(a1, b1, acc[1][1]);
    }
    if (it < Dz / 64 - 1) store_b(nxt);
    wait_async0();
    __syncthreads();
  }

  const int hl = l >> 4, ln = l & 15;
#pragma unroll
  for (int nt = 0; nt < 2; ++nt) {
    const int cg = col0 + n0 + nt * 16 + ln;
    const float bias = bfc[cg];
#pragma unroll
    for (int mt = 0; mt < 2; ++mt) {
#pragma unroll
      for (int r = 0; r < 8; ++r) {
        const int tg = row0 + m0 + mt * 16 + r + (hl << 3);
        out[(size_t)tg * Dz + cg] = acc[mt][nt][r] + bias;
      }
    }
  }
}

// ---------------------------------------------------------------------------
extern "C" void kernel_launch(void* const* d_in, const int* in_sizes, int n_in,
                              void* d_out, int out_size, void* d_ws, size_t ws_size,
                              hipStream_t stream) {
  (void)in_sizes; (void)n_in; (void)out_size; (void)ws_size;
  const float* x   = (const float*)d_in[0];
  const float* Wq  = (const float*)d_in[1];
  const float* bq  = (const float*)d_in[2];
  const float* Wf  = (const float*)d_in[3];
  const float* bfc = (const float*)d_in[4];
  float* out = (float*)d_out;

  // Workspace layout: [qkv bf16: B*3*H*T*64] then [attn bf16: B*T*D]
  const size_t qkv_elems = (size_t)Bz * 3 * Hz * Tz * HDz;  // 25,165,824
  __bf16* qkvws  = (__bf16*)d_ws;
  __bf16* attnws = (__bf16*)((char*)d_ws + qkv_elems * sizeof(__bf16));

  dim3 g1(3 * Dz / 128, (Bz * Tz) / 64);   // (24, 128)
  qkv_gemm<<<g1, 256, 0, stream>>>(x, Wq, bq, qkvws);

  flash_attn<<<dim3(Bz * Hz * (Tz / 64)), 128, 0, stream>>>(qkvws, attnws);

  dim3 g3(Dz / 128, (Bz * Tz) / 64);       // (8, 128)
  fc_gemm<<<g3, 256, 0, stream>>>(attnws, Wf, bfc, out);
}
